// Model_31284541784107
// MI455X (gfx1250) — matrix operand order repaired and checked by reading, compiled, not run on hardware
//
#include <hip/hip_runtime.h>
#include <hip/hip_bf16.h>

typedef __attribute__((ext_vector_type(16))) _Float16 v16h;
typedef __attribute__((ext_vector_type(8)))  _Float16 v8h;
typedef __attribute__((ext_vector_type(8)))  float    v8f;

static constexpr int   NN    = 10000;
static constexpr int   EE    = 320000;
static constexpr int   BB    = 262144;
static constexpr int   FNODE = 64;
static constexpr int   FEDGE = 8;
static constexpr int   HH    = 4;
static constexpr int   CCH   = 32;
static constexpr int   HC    = 128;          // H*C
static constexpr int   ET    = EE + NN;      // edges + self loops
static constexpr float SLOPE = 0.01f;
static constexpr float GSLOPE= 0.2f;
static constexpr float EPSV  = 1e-5f;

// ---------------- small utility kernels ----------------

__global__ void fill_u32_k(unsigned* p, long long n, unsigned v) {
  long long i = (long long)blockIdx.x * blockDim.x + threadIdx.x;
  long long stride = (long long)gridDim.x * blockDim.x;
  for (; i < n; i += stride) p[i] = v;
}

__global__ void cvt_w_k(const float* __restrict__ src, int ss,
                        _Float16* __restrict__ dst, int ds,
                        int rows, int cols) {
  long long tot = (long long)rows * cols;
  long long i = (long long)blockIdx.x * blockDim.x + threadIdx.x;
  long long stride = (long long)gridDim.x * blockDim.x;
  for (; i < tot; i += stride) {
    int r = (int)(i / cols), c = (int)(i % cols);
    dst[(long long)r * ds + c] = (_Float16)src[(long long)r * ss + c];
  }
}

__global__ void prep_od_k(const int* __restrict__ od, const float* __restrict__ distm,
                          const float* __restrict__ nv, float* __restrict__ extras) {
  int r = blockIdx.x * blockDim.x + threadIdx.x;
  if (r >= BB) return;
  int o = od[2 * r], d = od[2 * r + 1];
  extras[r * 5 + 0] = nv[o * 2 + 0];
  extras[r * 5 + 1] = nv[d * 2 + 0];
  extras[r * 5 + 2] = nv[o * 2 + 1];
  extras[r * 5 + 3] = nv[d * 2 + 1];
  extras[r * 5 + 4] = distm[(long long)o * NN + d];
}

// ---------------- GATv2 kernels ----------------

__device__ __forceinline__ unsigned encf(float x) {
  unsigned u = __float_as_uint(x);
  return (u & 0x80000000u) ? ~u : (u | 0x80000000u);
}
__device__ __forceinline__ float decf(unsigned u) {
  unsigned v = (u & 0x80000000u) ? (u & 0x7FFFFFFFu) : ~u;
  return __uint_as_float(v);
}

__global__ void gat_deg_k(const int* __restrict__ ei, const float* __restrict__ ea,
                          float* __restrict__ deg, float* __restrict__ ma) {
  int e = blockIdx.x * blockDim.x + threadIdx.x;
  if (e >= EE) return;
  int d = ei[EE + e];
  atomicAdd(&deg[d], 1.0f);
#pragma unroll
  for (int j = 0; j < FEDGE; j++) atomicAdd(&ma[d * FEDGE + j], ea[e * FEDGE + j]);
}

__global__ void gat_manorm_k(float* __restrict__ ma, const float* __restrict__ deg) {
  int i = blockIdx.x * blockDim.x + threadIdx.x;
  if (i >= NN * FEDGE) return;
  ma[i] /= fmaxf(deg[i / FEDGE], 1.0f);
}

__global__ void gat_lin_k(const float* __restrict__ x,
                          const float* __restrict__ Wl, const float* __restrict__ bl,
                          const float* __restrict__ Wr, const float* __restrict__ br,
                          float* __restrict__ xl, float* __restrict__ xr) {
  __shared__ float xs[FNODE];
  int n = blockIdx.x, t = threadIdx.x;   // blockDim = 128
  if (t < FNODE) xs[t] = x[n * FNODE + t];
  __syncthreads();
  float al = bl[t], ar = br[t];
  for (int k = 0; k < FNODE; k++) {
    float xv = xs[k];
    al += xv * Wl[t * FNODE + k];
    ar += xv * Wr[t * FNODE + k];
  }
  xl[(long long)n * HC + t] = al;
  xr[(long long)n * HC + t] = ar;
}

__global__ void gat_scr_k(const int* __restrict__ ei, const float* __restrict__ ea,
                          const float* __restrict__ ma, const float* __restrict__ We,
                          const float* __restrict__ att, const float* __restrict__ xl,
                          const float* __restrict__ xr, float* __restrict__ scr,
                          unsigned* __restrict__ mxenc) {
  int idx = blockIdx.x * blockDim.x + threadIdx.x;
  if (idx >= ET * HH) return;
  int h = idx & 3;
  int e = idx >> 2;
  int s, d; const float* eav;
  if (e < EE) { s = ei[e]; d = ei[EE + e]; eav = ea + (long long)e * FEDGE; }
  else        { s = d = e - EE;            eav = ma + (long long)(e - EE) * FEDGE; }
  float ev[FEDGE];
#pragma unroll
  for (int j = 0; j < FEDGE; j++) ev[j] = eav[j];
  float sc = 0.f;
  for (int c = 0; c < CCH; c++) {
    int oc = h * CCH + c;
    float em = 0.f;
#pragma unroll
    for (int j = 0; j < FEDGE; j++) em += ev[j] * We[oc * FEDGE + j];
    float m = xl[(long long)s * HC + oc] + xr[(long long)d * HC + oc] + em;
    m = m > 0.f ? m : GSLOPE * m;
    sc += m * att[oc];
  }
  scr[idx] = sc;
  atomicMax(&mxenc[d * HH + h], encf(sc));
}

__global__ void gat_exp_k(const int* __restrict__ ei, const float* __restrict__ scr,
                          const unsigned* __restrict__ mxenc, float* __restrict__ den,
                          float* __restrict__ alpha) {
  int idx = blockIdx.x * blockDim.x + threadIdx.x;
  if (idx >= ET * HH) return;
  int h = idx & 3;
  int e = idx >> 2;
  int d = (e < EE) ? ei[EE + e] : (e - EE);
  float mx = decf(mxenc[d * HH + h]);
  float ex = expf(scr[idx] - mx);
  alpha[idx] = ex;
  atomicAdd(&den[d * HH + h], ex);
}

__global__ void gat_agg_k(const int* __restrict__ ei, const float* __restrict__ den,
                          float* __restrict__ alpha, const float* __restrict__ xl,
                          float* __restrict__ acc) {
  int idx = blockIdx.x * blockDim.x + threadIdx.x;
  if (idx >= ET * HH) return;
  int h = idx & 3;
  int e = idx >> 2;
  int s, d;
  if (e < EE) { s = ei[e]; d = ei[EE + e]; } else { s = d = e - EE; }
  float a = alpha[idx] / den[d * HH + h];
  alpha[idx] = a;
  for (int c = 0; c < CCH; c++)
    atomicAdd(&acc[(long long)d * HC + h * CCH + c], a * xl[(long long)s * HC + h * CCH + c]);
}

__global__ void gat_fin_k(const float* __restrict__ acc, const float* __restrict__ bias,
                          _Float16* __restrict__ nf) {
  int i = blockIdx.x * blockDim.x + threadIdx.x;
  if (i >= NN * HC) return;
  float v = acc[i] + bias[i & (HC - 1)];
  nf[i] = (_Float16)fmaxf(v, 0.f);
}

// ---------------- WMMA GEMM kernels (register-blocked along N) ----------------
// Per-tile math: C(16x16,f32) += A(16x32,f16) x B(32x16,f16).
// A lane layout: row = lane&15; elems 0..15 -> K=k0+16*hi+j (same order as B)
// B lane layout: col = lane&15; elems 0..15 -> K=k0+16*hi+j  (B[k][col] = W[col][k])
// C lane layout: col = lane&15; VGPR i -> row = i + 8*hi
// One wave owns NT adjacent column tiles (16*NT cols) so each A fragment feeds NT WMMAs,
// and activations are re-read from HBM at most Nout/(16*NT) times per layer.
// K is a template parameter (== ldw for all layers) so B tile offsets are immediates.

template<int NT, int K>
__global__ void gemm_blk_k(const _Float16* __restrict__ A, int lda,
                           const _Float16* __restrict__ W,
                           int nColBlocks,
                           _Float16* __restrict__ outH, float* __restrict__ outF,
                           int ldo, float slope, int act) {
  int wave = blockIdx.x * (blockDim.x >> 5) + (threadIdx.x >> 5);
  int lane = threadIdx.x & 31;
  int tm = wave / nColBlocks;
  int tb = wave - tm * nColBlocks;
  if (tm * 16 >= BB) return;
  int hi = lane >> 4;
  int lr = lane & 15;
  const _Float16* arow = A + (long long)(tm * 16 + lr) * lda;
  int colBase = tb * NT * 16;
  const _Float16* wbase = W + (long long)(colBase + lr) * K + 16 * hi;
  v8f c[NT];
  v8f z = {};
#pragma unroll
  for (int t = 0; t < NT; t++) c[t] = z;
  union AV { v16h v; v8h h[2]; };
  for (int k0 = 0; k0 < K; k0 += 32) {
    AV a;
    a.h[0] = *(const v8h*)(arow + k0 + 16 * hi);
    a.h[1] = *(const v8h*)(arow + k0 + 16 * hi + 8);
#pragma unroll
    for (int t = 0; t < NT; t++) {
      v16h b = *(const v16h*)(wbase + (long long)t * 16 * K + k0);
      c[t] = __builtin_amdgcn_wmma_f32_16x16x32_f16(false, a.v, false, b, (short)0, c[t], false, false);
    }
  }
#pragma unroll
  for (int t = 0; t < NT; t++) {
    int col = colBase + t * 16 + lr;
#pragma unroll
    for (int i = 0; i < 8; i++) {
      int row = tm * 16 + i + 8 * hi;
      float v = c[t][i];
      if (act) v = v > 0.f ? v : v * slope;
      if (outH) outH[(long long)row * ldo + col] = (_Float16)v;
      if (outF) outF[(long long)row * ldo + col] = v;
    }
  }
}

// evo layer 1: A rows gathered from [n1[o] | n1[dn] | n2[o] | n2[dn]] (f16, 128 each),
// K=512 via WMMA; column 512 (dist) folded in as rank-1 epilogue. NT=8 column tiles.
__global__ void gemm_evo1_k(const _Float16* __restrict__ n1f, const _Float16* __restrict__ n2f,
                            const int* __restrict__ od, const float* __restrict__ extras,
                            const float* __restrict__ eW1, const _Float16* __restrict__ Wk,
                            _Float16* __restrict__ outH) {
  constexpr int NT = 8, K = 512;
  int wave = blockIdx.x * (blockDim.x >> 5) + (threadIdx.x >> 5);
  int lane = threadIdx.x & 31;
  const int nColBlocks = 2;            // 256 / (16*8)
  int tm = wave / nColBlocks;
  int tb = wave - tm * nColBlocks;
  if (tm * 16 >= BB) return;
  int hi = lane >> 4;
  int lr = lane & 15;
  int r = tm * 16 + lr;
  int o = od[2 * r], dn = od[2 * r + 1];
  int colBase = tb * NT * 16;
  const _Float16* wbase = Wk + (long long)(colBase + lr) * K + 16 * hi;
  v8f c[NT];
  v8f z = {};
#pragma unroll
  for (int t = 0; t < NT; t++) c[t] = z;
  union AV { v16h v; v8h h[2]; };
  for (int k0 = 0; k0 < K; k0 += 32) {
    AV a;
    int ka = k0 + 16 * hi;
    int kb = k0 + 16 * hi + 8;
    {
      int seg = ka >> 7;
      const _Float16* base = (seg < 2) ? n1f : n2f;
      int node = (seg & 1) ? dn : o;
      a.h[0] = *(const v8h*)(base + (long long)node * HC + (ka & 127));
    }
    {
      int seg = kb >> 7;
      const _Float16* base = (seg < 2) ? n1f : n2f;
      int node = (seg & 1) ? dn : o;
      a.h[1] = *(const v8h*)(base + (long long)node * HC + (kb & 127));
    }
#pragma unroll
    for (int t = 0; t < NT; t++) {
      v16h b = *(const v16h*)(wbase + (long long)t * 16 * K + k0);
      c[t] = __builtin_amdgcn_wmma_f32_16x16x32_f16(false, a.v, false, b, (short)0, c[t], false, false);
    }
  }
  float dist_i[8];
#pragma unroll
  for (int i = 0; i < 8; i++) dist_i[i] = extras[(tm * 16 + i + 8 * hi) * 5 + 4];
#pragma unroll
  for (int t = 0; t < NT; t++) {
    int col = colBase + t * 16 + lr;
    float wl = eW1[col * 513 + 512];
#pragma unroll
    for (int i = 0; i < 8; i++) {
      int row = tm * 16 + i + 8 * hi;
      float v = c[t][i] + dist_i[i] * wl;
      v = v > 0.f ? v : v * SLOPE;
      outH[(long long)row * 256 + col] = (_Float16)v;
    }
  }
}

// flow layer 1: A = evo (f32 in d_out, K=64, cvt->f16) + 5 extra cols as rank-5 epilogue.
__global__ void gemm_flow1_k(const float* __restrict__ evo, const float* __restrict__ extras,
                             const _Float16* __restrict__ Wk, const float* __restrict__ fW1,
                             _Float16* __restrict__ outH) {
  constexpr int NT = 8, K = 64;
  int wave = blockIdx.x * (blockDim.x >> 5) + (threadIdx.x >> 5);
  int lane = threadIdx.x & 31;
  const int nColBlocks = 2;            // 256 / (16*8)
  int tm = wave / nColBlocks;
  int tb = wave - tm * nColBlocks;
  if (tm * 16 >= BB) return;
  int hi = lane >> 4;
  int lr = lane & 15;
  int r = tm * 16 + lr;
  int colBase = tb * NT * 16;
  const float* erow = evo + (long long)r * 64;
  const _Float16* wbase = Wk + (long long)(colBase + lr) * K + 16 * hi;
  v8f c[NT];
  v8f z = {};
#pragma unroll
  for (int t = 0; t < NT; t++) c[t] = z;
  union AV { v16h v; _Float16 e[16]; };
#pragma unroll
  for (int k0 = 0; k0 < K; k0 += 32) {
    AV a;
    v8f f0 = *(const v8f*)(erow + k0 + 16 * hi);
    v8f f1 = *(const v8f*)(erow + k0 + 16 * hi + 8);
#pragma unroll
    for (int j = 0; j < 8; j++) { a.e[j] = (_Float16)f0[j]; a.e[8 + j] = (_Float16)f1[j]; }
#pragma unroll
    for (int t = 0; t < NT; t++) {
      v16h b = *(const v16h*)(wbase + (long long)t * 16 * K + k0);
      c[t] = __builtin_amdgcn_wmma_f32_16x16x32_f16(false, a.v, false, b, (short)0, c[t], false, false);
    }
  }
  float exv[8][5];
#pragma unroll
  for (int i = 0; i < 8; i++) {
    int row = tm * 16 + i + 8 * hi;
#pragma unroll
    for (int j = 0; j < 5; j++) exv[i][j] = extras[row * 5 + j];
  }
#pragma unroll
  for (int t = 0; t < NT; t++) {
    int col = colBase + t * 16 + lr;
    float wx[5];
#pragma unroll
    for (int j = 0; j < 5; j++) wx[j] = fW1[col * 69 + 64 + j];
#pragma unroll
    for (int i = 0; i < 8; i++) {
      int row = tm * 16 + i + 8 * hi;
      float v = c[t][i];
#pragma unroll
      for (int j = 0; j < 5; j++) v += exv[i][j] * wx[j];
      outH[(long long)row * 256 + col] = (_Float16)v;
    }
  }
}

// ---------------- batch norm + output kernels ----------------

__global__ void bn_stats_k(const _Float16* __restrict__ h, int cols,
                           float* __restrict__ sum, float* __restrict__ sq) {
  int c = threadIdx.x;                 // blockDim == cols
  float s = 0.f, s2 = 0.f;
  for (long long r = blockIdx.x; r < BB; r += gridDim.x) {
    float v = (float)h[r * cols + c];
    s += v; s2 += v * v;
  }
  atomicAdd(&sum[c], s);
  atomicAdd(&sq[c], s2);
}

__global__ void bn_norm_k(_Float16* __restrict__ h, int cols,
                          const float* __restrict__ sum, const float* __restrict__ sq,
                          const float* __restrict__ g, const float* __restrict__ b) {
  long long idx = (long long)blockIdx.x * blockDim.x + threadIdx.x;
  long long tot = (long long)BB * cols;
  if (idx >= tot) return;
  int c = (int)(idx % cols);
  float mu = sum[c] * (1.0f / BB);
  float var = sq[c] * (1.0f / BB) - mu * mu;
  float v = ((float)h[idx] - mu) * rsqrtf(var + EPSV) * g[c] + b[c];
  h[idx] = (_Float16)(v > 0.f ? v : v * SLOPE);
}

__global__ void flow_out_k(const _Float16* __restrict__ h, const float* __restrict__ w,
                           float* __restrict__ o) {
  int r = blockIdx.x * blockDim.x + threadIdx.x;
  if (r >= BB) return;
  const v8h* hr = (const v8h*)(h + (long long)r * 128);
  float acc = 0.f;
#pragma unroll
  for (int k = 0; k < 16; k++) {
    v8h hv = hr[k];
#pragma unroll
    for (int j = 0; j < 8; j++) acc += (float)hv[j] * w[k * 8 + j];
  }
  o[r] = acc;
}

// ---------------- host launch ----------------

extern "C" void kernel_launch(void* const* d_in, const int* in_sizes, int n_in,
                              void* d_out, int out_size, void* d_ws, size_t ws_size,
                              hipStream_t stream) {
  (void)in_sizes; (void)n_in; (void)out_size; (void)ws_size;
  const float* x1    = (const float*)d_in[0];
  const int*   ei1   = (const int*)  d_in[1];
  const float* ea1   = (const float*)d_in[2];
  const float* x2    = (const float*)d_in[3];
  const int*   ei2   = (const int*)  d_in[4];
  const float* ea2   = (const float*)d_in[5];
  const float* distm = (const float*)d_in[6];
  const float* nv    = (const float*)d_in[7];
  const int*   od    = (const int*)  d_in[8];
  const float* g1_Wl = (const float*)d_in[9];
  const float* g1_bl = (const float*)d_in[10];
  const float* g1_Wr = (const float*)d_in[11];
  const float* g1_br = (const float*)d_in[12];
  const float* g1_We = (const float*)d_in[13];
  const float* g1_att= (const float*)d_in[14];
  const float* g1_bias=(const float*)d_in[15];
  const float* g2_Wl = (const float*)d_in[16];
  const float* g2_bl = (const float*)d_in[17];
  const float* g2_Wr = (const float*)d_in[18];
  const float* g2_br = (const float*)d_in[19];
  const float* g2_We = (const float*)d_in[20];
  const float* g2_att= (const float*)d_in[21];
  const float* g2_bias=(const float*)d_in[22];
  const float* e_W1  = (const float*)d_in[23];
  const float* e_W2  = (const float*)d_in[24];
  const float* e_W3  = (const float*)d_in[25];
  const float* e_W4  = (const float*)d_in[26];
  const float* e_Wout= (const float*)d_in[27];
  const float* f_W1  = (const float*)d_in[28];
  const float* f_W2  = (const float*)d_in[29];
  const float* f_W3  = (const float*)d_in[30];
  const float* f_W4  = (const float*)d_in[31];
  const float* f_Wout= (const float*)d_in[32];
  const float* f_g1  = (const float*)d_in[33];
  const float* f_b1  = (const float*)d_in[34];
  const float* f_g2  = (const float*)d_in[35];
  const float* f_b2  = (const float*)d_in[36];
  const float* f_g3  = (const float*)d_in[37];
  const float* f_b3  = (const float*)d_in[38];
  const float* f_g4  = (const float*)d_in[39];
  const float* f_b4  = (const float*)d_in[40];

  float* out_inferred = (float*)d_out;
  float* out_a1  = out_inferred + BB;
  float* out_a2  = out_a1 + (long long)ET * HH;
  float* out_evo = out_a2 + (long long)ET * HH;

  // ---- workspace bump allocator ----
  char* wsp = (char*)d_ws;
  size_t off = 0;
  auto alloc = [&](size_t bytes) -> void* {
    void* p = wsp + off;
    off += (bytes + 255) & ~(size_t)255;
    return p;
  };
  _Float16* hA     = (_Float16*)alloc((size_t)BB * 256 * 2);
  _Float16* hB     = (_Float16*)alloc((size_t)BB * 256 * 2);
  float*    extras = (float*)   alloc((size_t)BB * 5 * 4);
  _Float16* n1f16  = (_Float16*)alloc((size_t)NN * HC * 2);
  _Float16* n2f16  = (_Float16*)alloc((size_t)NN * HC * 2);
  _Float16* eW1k   = (_Float16*)alloc((size_t)256 * 512 * 2);
  _Float16* eW2h   = (_Float16*)alloc((size_t)256 * 256 * 2);
  _Float16* eW3h   = (_Float16*)alloc((size_t)256 * 256 * 2);
  _Float16* eW4h   = (_Float16*)alloc((size_t)128 * 256 * 2);
  _Float16* eWouth = (_Float16*)alloc((size_t)64 * 128 * 2);
  _Float16* fW1k   = (_Float16*)alloc((size_t)256 * 64 * 2);
  _Float16* fW2h   = (_Float16*)alloc((size_t)256 * 256 * 2);
  _Float16* fW3h   = (_Float16*)alloc((size_t)256 * 256 * 2);
  _Float16* fW4h   = (_Float16*)alloc((size_t)128 * 256 * 2);
  float*    bnSum  = (float*)   alloc(256 * 4);
  float*    bnSq   = (float*)   alloc(256 * 4);
  // GAT scratch overlapped into hA (GAT completes before first GEMM writes hA)
  char* gp = (char*)hA;
  size_t goff = 0;
  auto galloc = [&](size_t bytes) -> void* {
    void* p = gp + goff;
    goff += (bytes + 255) & ~(size_t)255;
    return p;
  };
  float*    deg   = (float*)   galloc((size_t)NN * 4);
  float*    ma    = (float*)   galloc((size_t)NN * FEDGE * 4);
  float*    xl    = (float*)   galloc((size_t)NN * HC * 4);
  float*    xr    = (float*)   galloc((size_t)NN * HC * 4);
  float*    scr   = (float*)   galloc((size_t)ET * HH * 4);
  unsigned* mxenc = (unsigned*)galloc((size_t)NN * HH * 4);
  float*    den   = (float*)   galloc((size_t)NN * HH * 4);
  float*    acc   = (float*)   galloc((size_t)NN * HC * 4);

  auto fillu = [&](void* p, long long n, unsigned v) {
    int blocks = (int)((n + 255) / 256);
    if (blocks > 4096) blocks = 4096;
    fill_u32_k<<<blocks, 256, 0, stream>>>((unsigned*)p, n, v);
  };
  auto cvt = [&](const float* s, int ss, _Float16* d, int ds, int rows, int cols) {
    long long tot = (long long)rows * cols;
    int blocks = (int)((tot + 255) / 256);
    if (blocks > 4096) blocks = 4096;
    cvt_w_k<<<blocks, 256, 0, stream>>>(s, ss, d, ds, rows, cols);
  };

  // ---- prep ----
  prep_od_k<<<(BB + 255) / 256, 256, 0, stream>>>(od, distm, nv, extras);
  cvt(e_W1, 513, eW1k, 512, 256, 512);
  cvt(e_W2, 256, eW2h, 256, 256, 256);
  cvt(e_W3, 256, eW3h, 256, 256, 256);
  cvt(e_W4, 256, eW4h, 256, 128, 256);
  cvt(e_Wout, 128, eWouth, 128, 64, 128);
  cvt(f_W1, 69, fW1k, 64, 256, 64);
  cvt(f_W2, 256, fW2h, 256, 256, 256);
  cvt(f_W3, 256, fW3h, 256, 256, 256);
  cvt(f_W4, 256, fW4h, 256, 128, 256);

  // ---- GATv2 (sequential per graph, shared scratch) ----
  auto runGat = [&](const float* x, const int* ei, const float* ea,
                    const float* Wl, const float* bl, const float* Wr, const float* br,
                    const float* We, const float* att, const float* bias,
                    float* alphaOut, _Float16* nf16) {
    fillu(deg, NN, 0u);
    fillu(ma, (long long)NN * FEDGE, 0u);
    fillu(mxenc, (long long)NN * HH, 0x007FFFFFu);   // enc(-inf)
    fillu(den, (long long)NN * HH, 0u);
    fillu(acc, (long long)NN * HC, 0u);
    gat_deg_k<<<(EE + 255) / 256, 256, 0, stream>>>(ei, ea, deg, ma);
    gat_manorm_k<<<(NN * FEDGE + 255) / 256, 256, 0, stream>>>(ma, deg);
    gat_lin_k<<<NN, 128, 0, stream>>>(x, Wl, bl, Wr, br, xl, xr);
    int tot = ET * HH;
    gat_scr_k<<<(tot + 255) / 256, 256, 0, stream>>>(ei, ea, ma, We, att, xl, xr, scr, mxenc);
    gat_exp_k<<<(tot + 255) / 256, 256, 0, stream>>>(ei, scr, mxenc, den, alphaOut);
    gat_agg_k<<<(tot + 255) / 256, 256, 0, stream>>>(ei, den, alphaOut, xl, acc);
    gat_fin_k<<<(NN * HC + 255) / 256, 256, 0, stream>>>(acc, bias, nf16);
  };
  runGat(x1, ei1, ea1, g1_Wl, g1_bl, g1_Wr, g1_br, g1_We, g1_att, g1_bias, out_a1, n1f16);
  runGat(x2, ei2, ea2, g2_Wl, g2_bl, g2_Wr, g2_br, g2_We, g2_att, g2_bias, out_a2, n2f16);

  // ---- blocked WMMA GEMM launch helpers ----
  const int ROWTILES = BB / 16;   // 16384
  auto bn = [&](_Float16* h, int cols, const float* g, const float* b) {
    fillu(bnSum, cols, 0u);
    fillu(bnSq, cols, 0u);
    bn_stats_k<<<512, cols, 0, stream>>>(h, cols, bnSum, bnSq);
    long long tot = (long long)BB * cols;
    bn_norm_k<<<(int)((tot + 255) / 256), 256, 0, stream>>>(h, cols, bnSum, bnSq, g, b);
  };

  // ---- Evo MLP (f16 WMMA, leakyReLU fused) ----
  gemm_evo1_k<<<ROWTILES * 2 / 8, 256, 0, stream>>>(n1f16, n2f16, od, extras, e_W1, eW1k, hA);
  // 256 -> 256 (lrelu), NT=8, 2 col blocks
  gemm_blk_k<8, 256><<<ROWTILES * 2 / 8, 256, 0, stream>>>(hA, 256, eW2h, 2, hB, nullptr, 256, SLOPE, 1);
  gemm_blk_k<8, 256><<<ROWTILES * 2 / 8, 256, 0, stream>>>(hB, 256, eW3h, 2, hA, nullptr, 256, SLOPE, 1);
  // 256 -> 128 (lrelu), NT=8, 1 col block
  gemm_blk_k<8, 256><<<ROWTILES * 1 / 8, 256, 0, stream>>>(hA, 256, eW4h, 1, hB, nullptr, 128, SLOPE, 1);
  // 128 -> 64 (linear, f32 out), NT=4, 1 col block
  gemm_blk_k<4, 128><<<ROWTILES * 1 / 8, 256, 0, stream>>>(hB, 128, eWouth, 1, nullptr, out_evo, 64, SLOPE, 0);

  // ---- Flow MLP (WMMA GEMM + batch norm + leakyReLU) ----
  gemm_flow1_k<<<ROWTILES * 2 / 8, 256, 0, stream>>>(out_evo, extras, fW1k, f_W1, hA);
  bn(hA, 256, f_g1, f_b1);
  gemm_blk_k<8, 256><<<ROWTILES * 2 / 8, 256, 0, stream>>>(hA, 256, fW2h, 2, hB, nullptr, 256, SLOPE, 0);
  bn(hB, 256, f_g2, f_b2);
  gemm_blk_k<8, 256><<<ROWTILES * 2 / 8, 256, 0, stream>>>(hB, 256, fW3h, 2, hA, nullptr, 256, SLOPE, 0);
  bn(hA, 256, f_g3, f_b3);
  gemm_blk_k<8, 256><<<ROWTILES * 1 / 8, 256, 0, stream>>>(hA, 256, fW4h, 1, hB, nullptr, 128, SLOPE, 0);
  bn(hB, 128, f_g4, f_b4);
  flow_out_k<<<(BB + 255) / 256, 256, 0, stream>>>(hB, f_Wout, out_inferred);
}
